// SparseMoEBlock_80118319939670
// MI455X (gfx1250) — matrix operand order repaired and checked by reading, not compiled, not run on hardware
//
#include <hip/hip_runtime.h>
#include <math.h>

#define BB   4
#define CC   128
#define HH   128
#define WW   128
#define HW   (HH * WW)        // 16384
#define NTOK (BB * HW)        // 65536
#define EE   11
#define KACT 6
#define HID  512
#define NOUT (BB * CC * HH * WW)  // 8388608
#define MT   32               // tokens per wave tile (2 x 16-row WMMA subtiles)

typedef __attribute__((ext_vector_type(8)))  float        v8f;
typedef __attribute__((ext_vector_type(16))) __bf16       v16bf;
typedef __attribute__((ext_vector_type(4)))  unsigned int v4u;

struct Pack32 { v4u q0, q1; };  // 32 bytes = one 16-half bf16 fragment

__device__ __forceinline__ unsigned short f2bf(float f) {
  // round-to-nearest-even fp32 -> bf16
  unsigned int u = __float_as_uint(f);
  u += 0x7FFFu + ((u >> 16) & 1u);
  return (unsigned short)(u >> 16);
}

__device__ __forceinline__ v8f wmma_bf16(v16bf a, v16bf b, v8f c) {
  // D = A(16x32) * B(32x16) + C, f32 accumulate
  return __builtin_amdgcn_wmma_f32_16x16x32_bf16(
      /*neg_a=*/false, a, /*neg_b=*/false, b,
      /*c_mod=*/(short)0, c, /*reuse_a=*/false, /*reuse_b=*/false);
}

// Branchless tanh-form GELU: no EXEC manipulation, pure VALU
// (v_exp_f32 + v_rcp_f32). Error vs exact-erf GELU << bf16 rounding already
// introduced by the bf16 WMMA pipeline.
__device__ __forceinline__ float gelu_f(float v) {
  float u  = 0.7978845608028654f * (v + 0.044715f * v * v * v);
  float t  = __expf(2.0f * u);                           // branchless exp
  float th = 1.0f - 2.0f * __builtin_amdgcn_rcpf(t + 1.0f);  // tanh(u)
  return 0.5f * v * (1.0f + th);
}

// ---------------------------------------------------------------------------
// Kernel 1: fp32 -> bf16 weight conversion (one shot, tiny)
// ---------------------------------------------------------------------------
__global__ __launch_bounds__(256) void convert_w_kernel(
    const float* __restrict__ w1, const float* __restrict__ w2,
    unsigned short* __restrict__ w1b, unsigned short* __restrict__ w2b) {
  const int n = EE * HID * CC;  // 720896 (w1 and w2 are same size)
  int i = blockIdx.x * 256 + threadIdx.x;
  if (i < n)            w1b[i]     = f2bf(w1[i]);
  else if (i < 2 * n)   w2b[i - n] = f2bf(w2[i - n]);
}

// ---------------------------------------------------------------------------
// Kernel 2: router (gate + softmax + top-6 + mask) + channels-last bf16 x
//           + deterministic per-block aux-loss partial sums
// ---------------------------------------------------------------------------
__global__ __launch_bounds__(256) void router_kernel(
    const float* __restrict__ x, const float* __restrict__ gw,
    const float* __restrict__ gb, unsigned short* __restrict__ xb,
    float* __restrict__ mask, float* __restrict__ part) {
  __shared__ float red[256];
  const int tid  = threadIdx.x;
  const int t    = blockIdx.x * 256 + tid;      // token id
  const int bidx = t >> 14;                     // t / HW
  const int hw   = t & (HW - 1);

  float l[EE];
#pragma unroll
  for (int e = 0; e < EE; ++e) l[e] = gb[e];

  // coalesced along hw per channel; also emit channels-last bf16 copy of x
  for (int c = 0; c < CC; ++c) {
    float xv = x[((size_t)bidx * CC + c) * HW + hw];
    xb[(size_t)t * CC + c] = f2bf(xv);
#pragma unroll
    for (int e = 0; e < EE; ++e) l[e] = fmaf(xv, gw[e * CC + c], l[e]);
  }

  // softmax over experts
  float mx = l[0];
#pragma unroll
  for (int e = 1; e < EE; ++e) mx = fmaxf(mx, l[e]);
  float s = 0.f;
#pragma unroll
  for (int e = 0; e < EE; ++e) { l[e] = __expf(l[e] - mx); s += l[e]; }
  const float inv = 1.f / s;
#pragma unroll
  for (int e = 0; e < EE; ++e) l[e] *= inv;   // l[] now = scores

  // top-6 (first index wins ties, as in jax top_k), renormalize
  unsigned sel = 0; float psum = 0.f;
#pragma unroll
  for (int k = 0; k < KACT; ++k) {
    int best = 0; float bv = -1.f;
#pragma unroll
    for (int e = 0; e < EE; ++e) {
      float v = ((sel >> e) & 1u) ? -1.f : l[e];
      if (v > bv) { bv = v; best = e; }
    }
    sel |= 1u << best; psum += bv;
  }
  const float invp = 1.f / psum;
#pragma unroll
  for (int e = 0; e < EE; ++e)
    mask[(size_t)t * EE + e] = ((sel >> e) & 1u) ? l[e] * invp : 0.f;

  // deterministic block reductions: mean_prob and mean_load partials
#pragma unroll 1
  for (int e = 0; e < EE; ++e) {
    red[tid] = l[e];
    __syncthreads();
    for (int off = 128; off > 0; off >>= 1) {
      if (tid < off) red[tid] += red[tid + off];
      __syncthreads();
    }
    if (tid == 0) part[blockIdx.x * (2 * EE) + e] = red[0];
    __syncthreads();
    red[tid] = ((sel >> e) & 1u) ? 1.f : 0.f;
    __syncthreads();
    for (int off = 128; off > 0; off >>= 1) {
      if (tid < off) red[tid] += red[tid + off];
      __syncthreads();
    }
    if (tid == 0) part[blockIdx.x * (2 * EE) + EE + e] = red[0];
    __syncthreads();
  }
}

// ---------------------------------------------------------------------------
// Kernel 3: WMMA expert FFN. One wave32 per 32-token tile (2 M-subtiles).
//   Each loaded weight B-fragment feeds TWO independent WMMAs (no D->A/B
//   hazard, back-to-back XDL issue), halving L2 weight traffic vs M=16.
//   GEMM1: h = gelu(x[32x128] @ w1_e^T + b1)   (32 N-tiles x 4 K x 2 M WMMA)
//   GEMM2: o = h[32x512]  @ w2_e^T + b2        ( 8 N-tiles x 16 K x 2 M WMMA)
//   acc  += mask[t][e] * o  ;  y = x + acc (residual, NCHW store)
// ---------------------------------------------------------------------------
__global__ __launch_bounds__(32) void moe_ffn_kernel(
    const float* __restrict__ x, const unsigned short* __restrict__ xb,
    const unsigned short* __restrict__ w1b, const float* __restrict__ b1,
    const unsigned short* __restrict__ w2b, const float* __restrict__ b2,
    const float* __restrict__ mask, float* __restrict__ out) {
  __shared__ __attribute__((aligned(16))) unsigned short h_lds[MT * HID];  // 32KB
  __shared__ float smask[MT];

  const int lane = threadIdx.x;
  const int mrow = lane & 15;
  const int hi   = lane >> 4;       // half-wave id
  const int aoff = hi ? 8 : 0;      // A-fragment K sub-offset (ISA 7.12.2)
  const int boff = hi ? 8 : 0;      // B-fragment K sub-offset
  const int t0   = blockIdx.x * MT; // first token of tile

  // Preload A-fragments of the 32x128 x-tile (register-resident across experts)
  v16bf xa[2][4];
#pragma unroll
  for (int ms = 0; ms < 2; ++ms) {
    const unsigned short* xr = xb + (size_t)(t0 + ms * 16 + mrow) * CC;
#pragma unroll
    for (int kc = 0; kc < 4; ++kc) {
      Pack32 p;
      p.q0 = *(const v4u*)(xr + kc * 32 + aoff);
      p.q1 = *(const v4u*)(xr + kc * 32 + 16 + aoff);
      xa[ms][kc] = __builtin_bit_cast(v16bf, p);
    }
  }

  v8f oacc[2][8];
  {
    v8f z = {};
#pragma unroll
    for (int ms = 0; ms < 2; ++ms)
#pragma unroll
      for (int i = 0; i < 8; ++i) oacc[ms][i] = z;
  }

#pragma unroll 1
  for (int e = 0; e < EE; ++e) {
    __syncthreads();  // protect prev iteration's LDS reads
    smask[lane] = mask[(size_t)(t0 + lane) * EE + e];
    __syncthreads();

    float msum = 0.f;
#pragma unroll
    for (int i = 0; i < MT; ++i) msum += smask[i];
    if (msum == 0.f) continue;  // top-6 sparsity: skip expert for whole tile

    if (e + 1 < EE) {  // warm next expert's weights (global_prefetch_b8)
      __builtin_prefetch(w1b + (size_t)(e + 1) * HID * CC + lane * 64, 0, 0);
      __builtin_prefetch(w2b + (size_t)(e + 1) * CC * HID + lane * 64, 0, 0);
    }

    // ---- GEMM1 + bias + branchless GELU -> bf16 h tile in LDS ----
    const unsigned short* w1e = w1b + (size_t)e * HID * CC;
#pragma unroll 1
    for (int nt = 0; nt < 32; ++nt) {
      const int d0 = nt * 16;
      v8f acc0 = {}, acc1 = {};
      const unsigned short* wr = w1e + (size_t)(d0 + mrow) * CC + boff;
#pragma unroll
      for (int kc = 0; kc < 4; ++kc) {
        Pack32 p;
        p.q0 = *(const v4u*)(wr + kc * 32);
        p.q1 = *(const v4u*)(wr + kc * 32 + 16);
        v16bf bfr = __builtin_bit_cast(v16bf, p);
        acc0 = wmma_bf16(xa[0][kc], bfr, acc0);   // shared B, independent D
        acc1 = wmma_bf16(xa[1][kc], bfr, acc1);
      }
      const float bias = b1[e * HID + d0 + mrow];
#pragma unroll
      for (int v = 0; v < 8; ++v) {
        float h0 = gelu_f(acc0[v] + bias);
        float h1 = gelu_f(acc1[v] + bias);
        h_lds[(v + hi * 8) * HID + d0 + mrow]      = f2bf(h0);
        h_lds[(16 + v + hi * 8) * HID + d0 + mrow] = f2bf(h1);
      }
    }
    __syncthreads();

    // ---- GEMM2 + bias, mask-weighted accumulate ----
    const unsigned short* w2e = w2b + (size_t)e * CC * HID;
#pragma unroll 1
    for (int ct = 0; ct < 8; ++ct) {
      const int c0 = ct * 16;
      v8f acc0 = {}, acc1 = {};
      const unsigned short* wr = w2e + (size_t)(c0 + mrow) * HID + boff;
      const unsigned short* hr0 = h_lds + mrow * HID;
      const unsigned short* hr1 = h_lds + (16 + mrow) * HID;
#pragma unroll 4
      for (int kc = 0; kc < 16; ++kc) {
        const int d0 = kc * 32;
        Pack32 pb, pa0, pa1;
        pb.q0  = *(const v4u*)(wr + d0);
        pb.q1  = *(const v4u*)(wr + d0 + 16);
        pa0.q0 = *(const v4u*)(hr0 + d0 + aoff);        // ds_load_b128
        pa0.q1 = *(const v4u*)(hr0 + d0 + 16 + aoff);
        pa1.q0 = *(const v4u*)(hr1 + d0 + aoff);
        pa1.q1 = *(const v4u*)(hr1 + d0 + 16 + aoff);
        v16bf bfr = __builtin_bit_cast(v16bf, pb);
        acc0 = wmma_bf16(__builtin_bit_cast(v16bf, pa0), bfr, acc0);
        acc1 = wmma_bf16(__builtin_bit_cast(v16bf, pa1), bfr, acc1);
      }
      const float bias = b2[e * CC + c0 + mrow];
#pragma unroll
      for (int v = 0; v < 8; ++v) {
        oacc[0][ct][v] += smask[hi * 8 + v]      * (acc0[v] + bias);
        oacc[1][ct][v] += smask[16 + hi * 8 + v] * (acc1[v] + bias);
      }
    }
  }

  // ---- residual add + NCHW store ----
  const int bidx = t0 >> 14;
  const int hw0  = t0 & (HW - 1);
#pragma unroll
  for (int ms = 0; ms < 2; ++ms) {
#pragma unroll
    for (int ct = 0; ct < 8; ++ct) {
      const int c = ct * 16 + mrow;
      const size_t base = ((size_t)bidx * CC + c) * HW + hw0 + ms * 16;
#pragma unroll
      for (int v = 0; v < 8; ++v) {
        const int m = v + hi * 8;  // token row within subtile (C layout)
        out[base + m] = x[base + m] + oacc[ms][ct][v];
      }
    }
  }
}

// ---------------------------------------------------------------------------
// Kernel 4: deterministic aux-loss finalize
// ---------------------------------------------------------------------------
__global__ void finalize_kernel(const float* __restrict__ part,
                                float* __restrict__ out) {
  if (threadIdx.x == 0 && blockIdx.x == 0) {
    float aux = 0.f;
    for (int e = 0; e < EE; ++e) {
      float p = 0.f, ld = 0.f;
      for (int blk = 0; blk < NTOK / 256; ++blk) {
        p  += part[blk * (2 * EE) + e];
        ld += part[blk * (2 * EE) + EE + e];
      }
      aux += (p / (float)NTOK) * (ld / (float)NTOK);
    }
    out[NOUT] = (float)EE * aux;
  }
}

// ---------------------------------------------------------------------------
extern "C" void kernel_launch(void* const* d_in, const int* in_sizes, int n_in,
                              void* d_out, int out_size, void* d_ws,
                              size_t ws_size, hipStream_t stream) {
  const float* x  = (const float*)d_in[0];
  const float* gw = (const float*)d_in[1];
  const float* gb = (const float*)d_in[2];
  const float* w1 = (const float*)d_in[3];
  const float* b1 = (const float*)d_in[4];
  const float* w2 = (const float*)d_in[5];
  const float* b2 = (const float*)d_in[6];
  float* out = (float*)d_out;

  char* ws = (char*)d_ws;
  size_t off = 0;
  auto alloc = [&](size_t bytes) {
    char* p = ws + off;
    off = (off + bytes + 255) & ~(size_t)255;
    return p;
  };
  unsigned short* xb   = (unsigned short*)alloc((size_t)NTOK * CC * 2);      // 16.8 MB
  unsigned short* w1b  = (unsigned short*)alloc((size_t)EE * HID * CC * 2);  // 1.44 MB
  unsigned short* w2b  = (unsigned short*)alloc((size_t)EE * CC * HID * 2);  // 1.44 MB
  float*          mk   = (float*)alloc((size_t)NTOK * EE * 4);               // 2.88 MB
  float*          part = (float*)alloc((size_t)(NTOK / 256) * 2 * EE * 4);   // 22.5 KB

  hipLaunchKernelGGL(convert_w_kernel, dim3((2 * EE * HID * CC + 255) / 256),
                     dim3(256), 0, stream, w1, w2, w1b, w2b);
  hipLaunchKernelGGL(router_kernel, dim3(NTOK / 256), dim3(256), 0, stream,
                     x, gw, gb, xb, mk, part);
  hipLaunchKernelGGL(moe_ffn_kernel, dim3(NTOK / MT), dim3(32), 0, stream,
                     x, xb, w1b, b1, w2b, b2, mk, out);
  hipLaunchKernelGGL(finalize_kernel, dim3(1), dim3(1), 0, stream, part, out);
}
